// ComplexCustomCNN_53592601919917
// MI455X (gfx1250) — matrix operand order repaired and checked by reading, not compiled, not run on hardware
//
#include <hip/hip_runtime.h>

typedef _Float16 half_t;
typedef __attribute__((ext_vector_type(16))) _Float16 v16h;
typedef __attribute__((ext_vector_type(8)))  _Float16 v8h;
typedef __attribute__((ext_vector_type(8)))  float    v8f;
typedef __attribute__((ext_vector_type(4)))  unsigned int u32x4;
typedef __attribute__((ext_vector_type(8)))  int i32x8;
typedef __attribute__((ext_vector_type(4)))  int i32x4;

#define CEPS 1e-5f

// conv GEMM tiling
#define KS 64    // K per stage (2 wmma-K)
#define MT 32    // M tile per block
#define NT 128   // N tile per block
#define BPAD 72  // column stride (halfs) of column-major B tile (64 + 8 pad)

// Tensor Data Mover availability (device pass only; host pass parses fallback)
#if defined(__AMDGCN__) && defined(__has_builtin)
#if __has_builtin(__builtin_amdgcn_tensor_load_to_lds) && \
    __has_builtin(__builtin_amdgcn_s_wait_tensorcnt)
#define HAVE_TDM 1
#endif
#endif
#ifndef HAVE_TDM
#define HAVE_TDM 0
#endif

#if HAVE_TDM
// Issue a TDM DMA of a 2-D tile: MT rows x KS halfs, row stride kStride halfs,
// from global gsrc into LDS byte offset ldsoff (row-major, no padding).
// D# layout per CDNA5 ISA.
__device__ __forceinline__ void tdm_load_a(const half_t* gsrc, unsigned ldsoff,
                                           int kStride, int rows) {
  unsigned long long ga = (unsigned long long)(size_t)gsrc;
  u32x4 g0;
  g0[0] = 1u;                                        // count=1 (valid user D#)
  g0[1] = ldsoff;                                    // lds_addr (bytes)
  g0[2] = (unsigned)ga;                              // global_addr[31:0]
  g0[3] = (unsigned)((ga >> 32) & 0x01ffffffu)       // global_addr[56:32]
          | (2u << 30);                              // type = 2 ("image")
  i32x8 g1;
  g1[0] = (1 << 16);                                 // data_size = 1 -> 2 bytes
  g1[1] = (kStride & 0xffff) << 16;                  // tensor_dim0[15:0]
  g1[2] = (int)(((unsigned)kStride >> 16) |          // tensor_dim0[31:16]
                (((unsigned)rows & 0xffffu) << 16)); // tensor_dim1[15:0]
  g1[3] = (KS & 0xffff) << 16;                       // tensor_dim1 hi=0|tile_dim0
  g1[4] = (MT & 0xffff);                             // tile_dim1 (tile_dim2=0)
  g1[5] = kStride;                                   // tensor_dim0_stride[31:0]
  g1[6] = 0;                                         // stride hi | dim1_stride lo
  g1[7] = 0;
  i32x4 gz = {0, 0, 0, 0};
#if __clang_major__ >= 23
  i32x8 gz8 = {0, 0, 0, 0, 0, 0, 0, 0};
  __builtin_amdgcn_tensor_load_to_lds(g0, g1, gz, gz, gz8, 0);
#else
  __builtin_amdgcn_tensor_load_to_lds(g0, g1, gz, gz, 0);
#endif
}
#endif

// ---------------- WMMA fragment helpers (wave32, v_wmma_f32_16x16x32_f16) ----
__device__ __forceinline__ v16h load_a_frag(const half_t* A, int ld, int lane) {
  int m = lane & 15;
  int kbase = (lane & 16) ? 16 : 0;
  const half_t* row = A + m * ld + kbase;
  v8h lo = *(const v8h*)(row);
  v8h hi = *(const v8h*)(row + 8);
  return __builtin_shufflevector(lo, hi, 0, 1, 2, 3, 4, 5, 6, 7,
                                 8, 9, 10, 11, 12, 13, 14, 15);
}

__device__ __forceinline__ v16h load_b_frag(const half_t* B, int ld, int lane) {
  int n = lane & 15;
  int kbase = (lane & 16) ? 16 : 0;
  v16h b;
#pragma unroll
  for (int j = 0; j < 16; ++j) b[j] = B[(kbase + j) * ld + n];
  return b;
}

__device__ __forceinline__ v16h load_b_frag_cm(const half_t* Bc, int ldc,
                                               int lane) {
  int n = lane & 15;
  int kbase = (lane & 16) ? 16 : 0;
  const half_t* p = Bc + n * ldc + kbase;
  v8h lo = *(const v8h*)(p);
  v8h hi = *(const v8h*)(p + 8);
  return __builtin_shufflevector(lo, hi, 0, 1, 2, 3, 4, 5, 6, 7,
                                 8, 9, 10, 11, 12, 13, 14, 15);
}

// ---------------- prep kernels ----------------

__global__ __launch_bounds__(256) void prep_input_kernel(
    const float* __restrict__ x, half_t* __restrict__ act0,
    int Cin, int Bn, int HW) {
  int Ntot = Bn * HW;
  int total = 2 * Cin * Ntot;
  int idx = blockIdx.x * 256 + threadIdx.x;
  if (idx >= total) return;
  int c2 = idx / Ntot;
  int rem = idx - c2 * Ntot;
  int b = rem / HW;
  int q = rem - b * HW;
  float v = (c2 < Cin) ? x[((size_t)b * Cin + c2) * HW + q] : 0.f;
  act0[idx] = (half_t)v;
}

__global__ __launch_bounds__(256) void prep_conv_w_kernel(
    const float* __restrict__ wr, const float* __restrict__ wi,
    half_t* __restrict__ Wt, int Cout, int Cin, int Kp, int Kreal) {
  int total = 2 * Cout * Kp;
  int idx = blockIdx.x * 256 + threadIdx.x;
  if (idx >= total) return;
  int m = idx / Kp;
  int k = idx - m * Kp;
  float v = 0.f;
  if (k < Kreal) {
    int ci2 = k / 9;
    int r = k - ci2 * 9;
    bool isI = m >= Cout;
    int mm = isI ? (m - Cout) : m;
    bool ciI = ci2 >= Cin;
    int cc = ciI ? (ci2 - Cin) : ci2;
    size_t wo = ((size_t)mm * Cin + cc) * 9 + r;
    if (!isI) v = ciI ? -wi[wo] : wr[wo];
    else      v = ciI ?  wr[wo] : wi[wo];
  }
  Wt[idx] = (half_t)v;
}

__global__ __launch_bounds__(256) void prep_fc_w_kernel(
    const float* __restrict__ wr, const float* __restrict__ wi,
    half_t* __restrict__ Wt, int fo, int fi) {
  int K = 2 * fi, N = 2 * fo;
  int total = K * N;
  int idx = blockIdx.x * 256 + threadIdx.x;
  if (idx >= total) return;
  int k = idx / N;
  int n = idx - k * N;
  bool nI = n >= fo;
  int nn = nI ? (n - fo) : n;
  bool kI = k >= fi;
  int kk = kI ? (k - fi) : k;
  size_t wo = (size_t)nn * fi + kk;
  float v = nI ? (kI ? wr[wo] : wi[wo]) : (kI ? -wi[wo] : wr[wo]);
  Wt[idx] = (half_t)v;
}

__global__ __launch_bounds__(256) void combine_bias_kernel(
    const float* __restrict__ br, const float* __restrict__ bi,
    float* __restrict__ cb, int n) {
  int idx = blockIdx.x * 256 + threadIdx.x;
  if (idx >= 2 * n) return;
  cb[idx] = (idx < n) ? br[idx] : bi[idx - n];
}

// ---------------- implicit-GEMM conv via WMMA ----------------
// grid: (Ntot/NT, Cout2/MT), block 256 (8 waves, 2x4 wave grid).
// Weight (A) tiles are DMAed to LDS by the Tensor Data Mover (TENSORcnt),
// im2col B tiles are gathered into registers then stored column-major.
__global__ __launch_bounds__(256) void conv_wmma_kernel(
    const half_t* __restrict__ Wt,   // [Cout2][Kp] packed weights (A)
    const half_t* __restrict__ act,  // [Cin2][B*H*W] activations
    half_t* __restrict__ Y,          // [Cout2][B*H*W] conv out (pre-BN)
    const float* __restrict__ cbias, // [Cout2]
    int Kp, int Kreal, int Bn, int H, int Wd, int Ntot, int Cout2) {
  __shared__ __align__(16) half_t As[2][MT * KS];    // [buf][m][k] row-major
  __shared__ __align__(16) half_t Bs[2][NT * BPAD];  // [buf][n][k+pad] col-major
  const int tid = threadIdx.x;
  const int lane = tid & 31;
  const int wid = tid >> 5;
  const int wm = wid >> 2;
  const int wn = wid & 3;
  const int n0 = blockIdx.x * NT;
  const int m0 = blockIdx.y * MT;
  const int HW = H * Wd;

  const int am = tid >> 3;
  const int ak = (tid & 7) * 8;
  const half_t* wsrc = Wt + (size_t)(m0 + am) * Kp + ak;
  (void)wsrc;
  const int bn_ = tid >> 1;
  const int bh = (tid & 1) * 32;
  const int p = n0 + bn_;
  const int pb = p / HW;
  const int prem = p - pb * HW;
  const int ph = prem / Wd;
  const int pw = prem - ph * Wd;
  const half_t* arow = act + (size_t)pb * HW;

  const int nStages = Kp / KS;
  v8f acc0 = {}, acc1 = {};

  auto fetchB = [&](int k0, half_t* tmp) {
#pragma unroll
    for (int j = 0; j < 32; ++j) {
      int k = k0 + bh + j;
      half_t v = (half_t)0.f;
      if (k < Kreal) {
        int ci2 = k / 9;
        int r = k - ci2 * 9;
        int kh = r / 3;
        int kw = r - kh * 3;
        int ih = ph + kh - 1, iw = pw + kw - 1;
        if (ih >= 0 && ih < H && iw >= 0 && iw < Wd)
          v = arow[(size_t)ci2 * Ntot + ih * Wd + iw];
      }
      tmp[j] = v;
    }
  };
  auto storeB = [&](int buf, const half_t* tmp) {
#pragma unroll
    for (int q = 0; q < 4; ++q)
      *(v8h*)(&Bs[buf][bn_ * BPAD + bh + q * 8]) = *(const v8h*)(tmp + q * 8);
  };
  auto mma = [&](int buf) {
    const half_t* Ab = &As[buf][wm * 16 * KS];
    const half_t* Bb = &Bs[buf][(wn * 32) * BPAD];
    v16h a0 = load_a_frag(Ab, KS, lane);
    v16h a1 = load_a_frag(Ab + 32, KS, lane);
    v16h b00 = load_b_frag_cm(Bb, BPAD, lane);
    v16h b01 = load_b_frag_cm(Bb + 32, BPAD, lane);
    v16h b10 = load_b_frag_cm(Bb + 16 * BPAD, BPAD, lane);
    v16h b11 = load_b_frag_cm(Bb + 16 * BPAD + 32, BPAD, lane);
    acc0 = __builtin_amdgcn_wmma_f32_16x16x32_f16(false, a0, false, b00,
                                                  (short)0, acc0, false, false);
    acc0 = __builtin_amdgcn_wmma_f32_16x16x32_f16(false, a1, false, b01,
                                                  (short)0, acc0, false, false);
    acc1 = __builtin_amdgcn_wmma_f32_16x16x32_f16(false, a0, false, b10,
                                                  (short)0, acc1, false, false);
    acc1 = __builtin_amdgcn_wmma_f32_16x16x32_f16(false, a1, false, b11,
                                                  (short)0, acc1, false, false);
  };

  {  // prologue: stage 0
    half_t tmp[32] __attribute__((aligned(16)));
#if HAVE_TDM
    if (wid == 0)
      tdm_load_a(Wt + (size_t)m0 * Kp, (unsigned)(size_t)&As[0][0], Kp, Cout2);
#else
    v8h a = *(const v8h*)(wsrc);
#endif
    fetchB(0, tmp);
    storeB(0, tmp);
#if HAVE_TDM
    if (wid == 0) __builtin_amdgcn_s_wait_tensorcnt(0);
#else
    *(v8h*)(&As[0][am * KS + ak]) = a;
#endif
  }
  __syncthreads();
  for (int s = 0; s < nStages; ++s) {
    int buf = s & 1;
    if (s + 1 < nStages) {
      half_t tmp[32] __attribute__((aligned(16)));
#if HAVE_TDM
      if (wid == 0)
        tdm_load_a(Wt + (size_t)m0 * Kp + (s + 1) * KS,
                   (unsigned)(size_t)&As[buf ^ 1][0], Kp, Cout2);
#else
      if (s + 2 < nStages)
        __builtin_prefetch(wsrc + (s + 2) * KS, 0, 1);
      v8h a = *(const v8h*)(wsrc + (s + 1) * KS);
#endif
      fetchB((s + 1) * KS, tmp);
      mma(buf);
      storeB(buf ^ 1, tmp);
#if HAVE_TDM
      if (wid == 0) __builtin_amdgcn_s_wait_tensorcnt(0);
#else
      *(v8h*)(&As[buf ^ 1][am * KS + ak]) = a;
#endif
    } else {
      mma(buf);
    }
    __syncthreads();
  }

  const int rbase = (lane & 16) ? 8 : 0;
  const int c0 = n0 + wn * 32 + (lane & 15);
#pragma unroll
  for (int r = 0; r < 8; ++r) {
    int m = m0 + wm * 16 + rbase + r;
    float bias = cbias[m];
    Y[(size_t)m * Ntot + c0] = (half_t)(acc0[r] + bias);
    Y[(size_t)m * Ntot + c0 + 16] = (half_t)(acc1[r] + bias);
  }
}

// ---------------- complex BN statistics -> fused affine coefficients --------
__global__ __launch_bounds__(256) void bn_stats_kernel(
    const half_t* __restrict__ Y, const float* __restrict__ bnw,
    const float* __restrict__ bnb, float* __restrict__ coef,
    int Cout, int Ntot) {
  __shared__ float red[5][256];
  const int c = blockIdx.x;
  const int tid = threadIdx.x;
  const half_t* yr = Y + (size_t)c * Ntot;
  const half_t* yi = Y + (size_t)(c + Cout) * Ntot;
  float s0 = 0.f, s1 = 0.f, s2 = 0.f, s3 = 0.f, s4 = 0.f;
  for (int j = tid * 8; j < Ntot; j += 256 * 8) {
    v8h xr8 = *(const v8h*)(yr + j);
    v8h xi8 = *(const v8h*)(yi + j);
#pragma unroll
    for (int e = 0; e < 8; ++e) {
      float xr = (float)xr8[e], xi = (float)xi8[e];
      s0 += xr; s1 += xi; s2 += xr * xr; s3 += xi * xi; s4 += xr * xi;
    }
  }
  red[0][tid] = s0; red[1][tid] = s1; red[2][tid] = s2;
  red[3][tid] = s3; red[4][tid] = s4;
  __syncthreads();
  for (int s = 128; s > 0; s >>= 1) {
    if (tid < s) {
#pragma unroll
      for (int k = 0; k < 5; ++k) red[k][tid] += red[k][tid + s];
    }
    __syncthreads();
  }
  if (tid == 0) {
    float inv = 1.f / (float)Ntot;
    float mr = red[0][0] * inv, mi = red[1][0] * inv;
    float vrr = red[2][0] * inv - mr * mr + CEPS;
    float vii = red[3][0] * inv - mi * mi + CEPS;
    float vri = red[4][0] * inv - mr * mi;
    float s = sqrtf(vrr * vii - vri * vri);
    float t = sqrtf(vrr + vii + 2.f * s);
    float ist = 1.f / (s * t);
    float rrr = (vii + s) * ist;
    float rii = (vrr + s) * ist;
    float rri = -vri * ist;
    float wrr = bnw[c * 3 + 0], wii = bnw[c * 3 + 1], wri = bnw[c * 3 + 2];
    float br_ = bnb[c * 2 + 0], bi_ = bnb[c * 2 + 1];
    coef[c * 8 + 0] = wrr * rrr + wri * rri;
    coef[c * 8 + 1] = wrr * rri + wri * rii;
    coef[c * 8 + 2] = wri * rrr + wii * rri;
    coef[c * 8 + 3] = wri * rri + wii * rii;
    coef[c * 8 + 4] = mr;
    coef[c * 8 + 5] = mi;
    coef[c * 8 + 6] = br_;
    coef[c * 8 + 7] = bi_;
  }
}

// ---------------- BN apply + CReLU + magnitude-argmax 2x2 pool --------------
__global__ __launch_bounds__(256) void bn_relu_pool_kernel(
    const half_t* __restrict__ Y, const float* __restrict__ coef,
    half_t* __restrict__ out, int Cout, int Bn, int H, int Wd) {
  int Ho = H >> 1, Wo = Wd >> 1;
  int Nout = Bn * Ho * Wo;
  int total = Cout * Nout;
  int idx = blockIdx.x * 256 + threadIdx.x;
  if (idx >= total) return;
  int c = idx / Nout;
  int rem = idx - c * Nout;
  int b = rem / (Ho * Wo);
  int q = rem - b * (Ho * Wo);
  int oh = q / Wo;
  int ow = q - oh * Wo;
  const float* cf = coef + c * 8;
  float Arr = cf[0], Ari = cf[1], Air = cf[2], Aii = cf[3];
  float mr = cf[4], mi = cf[5], br_ = cf[6], bi_ = cf[7];
  int Ntot = Bn * H * Wd, HW = H * Wd;
  float bestm = -1.f, bR = 0.f, bI = 0.f;
#pragma unroll
  for (int d = 0; d < 4; ++d) {
    int h = 2 * oh + (d >> 1), w = 2 * ow + (d & 1);
    size_t o = (size_t)b * HW + h * Wd + w;
    float xr = (float)Y[(size_t)c * Ntot + o];
    float xi = (float)Y[(size_t)(c + Cout) * Ntot + o];
    float ypr = fmaxf(Arr * (xr - mr) + Ari * (xi - mi) + br_, 0.f);
    float ypi = fmaxf(Air * (xr - mr) + Aii * (xi - mi) + bi_, 0.f);
    float m2 = ypr * ypr + ypi * ypi;
    if (m2 > bestm) { bestm = m2; bR = ypr; bI = ypi; }
  }
  out[(size_t)c * Nout + rem] = (half_t)bR;
  out[(size_t)(c + Cout) * Nout + rem] = (half_t)bI;
}

__global__ __launch_bounds__(256) void bn_relu_kernel(
    const half_t* __restrict__ Y, const float* __restrict__ coef,
    half_t* __restrict__ out, int Cout, int Ntot) {
  int total = Cout * Ntot;
  int idx = blockIdx.x * 256 + threadIdx.x;
  if (idx >= total) return;
  int c = idx / Ntot;
  int rem = idx - c * Ntot;
  const float* cf = coef + c * 8;
  float xr = (float)Y[(size_t)c * Ntot + rem];
  float xi = (float)Y[(size_t)(c + Cout) * Ntot + rem];
  float ypr = fmaxf(cf[0] * (xr - cf[4]) + cf[1] * (xi - cf[5]) + cf[6], 0.f);
  float ypi = fmaxf(cf[2] * (xr - cf[4]) + cf[3] * (xi - cf[5]) + cf[7], 0.f);
  out[(size_t)c * Ntot + rem] = (half_t)ypr;
  out[(size_t)(c + Cout) * Ntot + rem] = (half_t)ypi;
}

__global__ __launch_bounds__(256) void avgpool_kernel(
    const half_t* __restrict__ act, half_t* __restrict__ A0,
    int C2, int Bn, int S) {
  int total = Bn * C2;
  int idx = blockIdx.x * 256 + threadIdx.x;
  if (idx >= total) return;
  int b = idx / C2;
  int c = idx - b * C2;
  const half_t* src = act + (size_t)c * (Bn * S) + b * S;
  float s = 0.f;
  for (int j = 0; j < S; j += 8) {
    v8h v = *(const v8h*)(src + j);
#pragma unroll
    for (int e = 0; e < 8; ++e) s += (float)v[e];
  }
  A0[(size_t)b * C2 + c] = (half_t)(s / (float)S);
}

// ---------------- FC GEMM via WMMA: one wave per 16x16 tile -----------------
__global__ __launch_bounds__(32) void fc_wmma_kernel(
    const half_t* __restrict__ A, const half_t* __restrict__ Wt,
    const float* __restrict__ cbias, half_t* __restrict__ outH,
    float* __restrict__ outF, int N, int K, int relu, int nstore) {
  const int lane = threadIdx.x;
  const int n0 = blockIdx.x * 16;
  const int m0 = blockIdx.y * 16;
  v8f acc = {};
  for (int k0 = 0; k0 < K; k0 += 32) {
    v16h af = load_a_frag(A + (size_t)m0 * K + k0, K, lane);
    v16h bf = load_b_frag(Wt + (size_t)k0 * N + n0, N, lane);
    acc = __builtin_amdgcn_wmma_f32_16x16x32_f16(false, af, false, bf,
                                                 (short)0, acc, false, false);
  }
  const int col = n0 + (lane & 15);
  const int rbase = (lane & 16) ? 8 : 0;
#pragma unroll
  for (int r = 0; r < 8; ++r) {
    int m = m0 + rbase + r;
    float y = acc[r] + cbias[col];
    if (relu) y = fmaxf(y, 0.f);
    if (col < nstore) {
      if (outH) outH[(size_t)m * N + col] = (half_t)y;
      else outF[(size_t)m * nstore + col] = y;
    }
  }
}

// ---------------- host orchestration ----------------

static void run_conv_layer(const float* wr, const float* wi, const float* br,
                           const float* bi, const float* bnw, const float* bnb,
                           const half_t* actIn, half_t* Wt, float* cb,
                           half_t* Y, float* coef, half_t* actOut,
                           int Cin, int Cout, int H, int Wd, bool pool,
                           hipStream_t stream) {
  const int Bn = 64;
  int Ntot = Bn * H * Wd;
  int Kreal = 2 * Cin * 9;
  int Kp = (Kreal + 63) & ~63;
  int Cout2 = 2 * Cout;
  prep_conv_w_kernel<<<(Cout2 * Kp + 255) / 256, 256, 0, stream>>>(
      wr, wi, Wt, Cout, Cin, Kp, Kreal);
  combine_bias_kernel<<<(Cout2 + 255) / 256, 256, 0, stream>>>(br, bi, cb, Cout);
  dim3 grid(Ntot / NT, Cout2 / MT);
  conv_wmma_kernel<<<grid, 256, 0, stream>>>(Wt, actIn, Y, cb, Kp, Kreal, Bn, H,
                                             Wd, Ntot, Cout2);
  bn_stats_kernel<<<Cout, 256, 0, stream>>>(Y, bnw, bnb, coef, Cout, Ntot);
  if (pool) {
    int total = Cout * Bn * (H / 2) * (Wd / 2);
    bn_relu_pool_kernel<<<(total + 255) / 256, 256, 0, stream>>>(
        Y, coef, actOut, Cout, Bn, H, Wd);
  } else {
    int total = Cout * Ntot;
    bn_relu_kernel<<<(total + 255) / 256, 256, 0, stream>>>(Y, coef, actOut,
                                                            Cout, Ntot);
  }
}

extern "C" void kernel_launch(void* const* d_in, const int* in_sizes, int n_in,
                              void* d_out, int out_size, void* d_ws,
                              size_t ws_size, hipStream_t stream) {
  (void)in_sizes; (void)n_in; (void)out_size; (void)ws_size;
  const float* x = (const float*)d_in[0];
  const float *cwr[4], *cwi[4], *cbr[4], *cbi[4], *bnw[4], *bnb[4];
  for (int i = 0; i < 4; ++i) {
    int base = 1 + 6 * i;
    cwr[i] = (const float*)d_in[base + 0];
    cwi[i] = (const float*)d_in[base + 1];
    cbr[i] = (const float*)d_in[base + 2];
    cbi[i] = (const float*)d_in[base + 3];
    bnw[i] = (const float*)d_in[base + 4];
    bnb[i] = (const float*)d_in[base + 5];
  }
  const float *fwr[3], *fwi[3], *fbr[3], *fbi[3];
  for (int i = 0; i < 3; ++i) {
    int base = 25 + 4 * i;
    fwr[i] = (const float*)d_in[base + 0];
    fwi[i] = (const float*)d_in[base + 1];
    fbr[i] = (const float*)d_in[base + 2];
    fbi[i] = (const float*)d_in[base + 3];
  }
  float* out = (float*)d_out;

  size_t off = 0;
  auto alloc = [&](size_t bytes) -> void* {
    void* p = (void*)((char*)d_ws + off);
    off += (bytes + 255) & ~(size_t)255;
    return p;
  };
  half_t* actA = (half_t*)alloc((size_t)128 * 65536 * 2);
  half_t* actB = (half_t*)alloc((size_t)128 * 65536 * 2);
  half_t* Y = (half_t*)alloc((size_t)128 * 262144 * 2);
  const int Kp[4] = {64, 1152, 2304, 4608};
  const int Co2[4] = {128, 256, 512, 1024};
  half_t* Wc[4];
  float* cbc[4];
  for (int i = 0; i < 4; ++i) {
    Wc[i] = (half_t*)alloc((size_t)Co2[i] * Kp[i] * 2);
    cbc[i] = (float*)alloc((size_t)Co2[i] * 4);
  }
  float* coef = (float*)alloc((size_t)512 * 8 * 4);
  half_t* A0 = (half_t*)alloc((size_t)64 * 1024 * 2);
  half_t* A1 = (half_t*)alloc((size_t)64 * 4096 * 2);
  half_t* A2 = (half_t*)alloc((size_t)64 * 1024 * 2);
  half_t* Wf0 = (half_t*)alloc((size_t)1024 * 4096 * 2);
  half_t* Wf1 = (half_t*)alloc((size_t)4096 * 1024 * 2);
  half_t* Wf2 = (half_t*)alloc((size_t)1024 * 20 * 2);
  float* cbf0 = (float*)alloc((size_t)4096 * 4);
  float* cbf1 = (float*)alloc((size_t)1024 * 4);
  float* cbf2 = (float*)alloc((size_t)20 * 4);

  prep_input_kernel<<<(6 * 262144 + 255) / 256, 256, 0, stream>>>(x, actA, 3,
                                                                  64, 4096);
  run_conv_layer(cwr[0], cwi[0], cbr[0], cbi[0], bnw[0], bnb[0], actA, Wc[0],
                 cbc[0], Y, coef, actB, 3, 64, 64, 64, true, stream);
  run_conv_layer(cwr[1], cwi[1], cbr[1], cbi[1], bnw[1], bnb[1], actB, Wc[1],
                 cbc[1], Y, coef, actA, 64, 128, 32, 32, true, stream);
  run_conv_layer(cwr[2], cwi[2], cbr[2], cbi[2], bnw[2], bnb[2], actA, Wc[2],
                 cbc[2], Y, coef, actB, 128, 256, 16, 16, true, stream);
  run_conv_layer(cwr[3], cwi[3], cbr[3], cbi[3], bnw[3], bnb[3], actB, Wc[3],
                 cbc[3], Y, coef, actA, 256, 512, 8, 8, false, stream);
  avgpool_kernel<<<(64 * 1024 + 255) / 256, 256, 0, stream>>>(actA, A0, 1024,
                                                              64, 64);
  prep_fc_w_kernel<<<(1024 * 4096 + 255) / 256, 256, 0, stream>>>(fwr[0],
                                                                  fwi[0], Wf0,
                                                                  2048, 512);
  combine_bias_kernel<<<(4096 + 255) / 256, 256, 0, stream>>>(fbr[0], fbi[0],
                                                              cbf0, 2048);
  fc_wmma_kernel<<<dim3(256, 4), 32, 0, stream>>>(A0, Wf0, cbf0, A1, nullptr,
                                                  4096, 1024, 1, 4096);
  prep_fc_w_kernel<<<(4096 * 1024 + 255) / 256, 256, 0, stream>>>(fwr[1],
                                                                  fwi[1], Wf1,
                                                                  512, 2048);
  combine_bias_kernel<<<(1024 + 255) / 256, 256, 0, stream>>>(fbr[1], fbi[1],
                                                              cbf1, 512);
  fc_wmma_kernel<<<dim3(64, 4), 32, 0, stream>>>(A1, Wf1, cbf1, A2, nullptr,
                                                 1024, 4096, 1, 1024);
  prep_fc_w_kernel<<<(1024 * 20 + 255) / 256, 256, 0, stream>>>(fwr[2], fwi[2],
                                                                Wf2, 10, 512);
  combine_bias_kernel<<<(20 + 255) / 256, 256, 0, stream>>>(fbr[2], fbi[2],
                                                            cbf2, 10);
  fc_wmma_kernel<<<dim3(1, 4), 32, 0, stream>>>(A2, Wf2, cbf2, nullptr, out,
                                                20, 1024, 0, 10);
}
